// YoloHeadsTaskAlignedAssigner_64252710748540
// MI455X (gfx1250) — compile-verified, matrix operand order checked by reading, not run on hardware
//
#include <hip/hip_runtime.h>
#include <hip/hip_bf16.h>
#include <stdint.h>

#define BB   16      // batch
#define LL   8400    // anchors
#define NGT  32      // gt per image
#define KPT  68      // keypoints
#define CC   1       // classes
#define TOPKK 13
#define EPSF 1e-9f

typedef __attribute__((ext_vector_type(16))) _Float16 v16h;
typedef __attribute__((ext_vector_type(8)))  float    v8f;

// ---------------------------------------------------------------- K0: zero mask
__global__ void k0_zero(unsigned* mask32, int count) {
    int i = blockIdx.x * blockDim.x + threadIdx.x;
    if (i < count) mask32[i] = 0u;
}

// ---------------------------------------------------------------- K1: IoU(gt, pred)
__global__ void k1_iou(const float* __restrict__ gt_bboxes,
                       const float* __restrict__ pred_bboxes,
                       float* __restrict__ iou) {
    int bi = blockIdx.y;                       // b*NGT + i
    int b  = bi / NGT;
    int l  = blockIdx.x * 256 + threadIdx.x;
    if (l >= LL) return;
    float4 g = *(const float4*)(gt_bboxes + (size_t)bi * 4);
    float4 p = *(const float4*)(pred_bboxes + ((size_t)b * LL + l) * 4);
    float ix0 = fmaxf(g.x, p.x), iy0 = fmaxf(g.y, p.y);
    float ix1 = fminf(g.z, p.z), iy1 = fminf(g.w, p.w);
    float ov  = fmaxf(ix1 - ix0, 0.f) * fmaxf(iy1 - iy0, 0.f);
    float a1  = fmaxf(g.z - g.x, 0.f) * fmaxf(g.w - g.y, 0.f);
    float a2  = fmaxf(p.z - p.x, 0.f) * fmaxf(p.w - p.y, 0.f);
    iou[(size_t)bi * LL + l] = ov / (a1 + a2 - ov + EPSF);
}

// ---------------------------------------------------------------- K2: top-13 per (b,i)
__global__ void k2_topk(const float* __restrict__ iou,
                        const float* __restrict__ pred_scores,
                        const float* __restrict__ anchor_points,
                        const float* __restrict__ gt_bboxes,
                        const float* __restrict__ pad_mask,
                        const int*   __restrict__ gt_labels,
                        unsigned*    __restrict__ mask32) {
    __shared__ float sm[LL];
    __shared__ float rv[256];
    __shared__ int   ri[256];
    int bi = blockIdx.x;                      // b*NGT + i
    int b  = bi / NGT;
    int i  = bi % NGT;
    float pad = pad_mask[bi];
    float gx0 = gt_bboxes[bi*4+0], gy0 = gt_bboxes[bi*4+1];
    float gx1 = gt_bboxes[bi*4+2], gy1 = gt_bboxes[bi*4+3];
    int   lbl = gt_labels[bi];
    const float* iourow = iou + (size_t)bi * LL;
    __builtin_prefetch(iourow, 0, 0);         // global_prefetch_b8
    for (int l = threadIdx.x; l < LL; l += blockDim.x) {
        float v  = iourow[l];
        float ax = anchor_points[l*2], ay = anchor_points[l*2+1];
        float dmin = fminf(fminf(ax - gx0, ay - gy0), fminf(gx1 - ax, gy1 - ay));
        float ingts = (dmin > EPSF) ? 1.f : 0.f;
        float s  = pred_scores[((size_t)b * LL + l) * CC + lbl];
        float v2 = v * v;
        sm[l] = s * v2 * v2 * v2 * ingts;     // score^ALPHA * iou^BETA * in_gts
    }
    __syncthreads();
    if (pad > 0.f) {                          // pad is uniform per block
        for (int t = 0; t < TOPKK; ++t) {
            float bv = -INFINITY; int bidx = 0x7fffffff;
            for (int l = threadIdx.x; l < LL; l += blockDim.x) {
                float v = sm[l];
                if (v > bv || (v == bv && l < bidx)) { bv = v; bidx = l; }
            }
            rv[threadIdx.x] = bv; ri[threadIdx.x] = bidx;
            __syncthreads();
            for (int s2 = 128; s2 > 0; s2 >>= 1) {
                if ((int)threadIdx.x < s2) {
                    float ov = rv[threadIdx.x + s2]; int oi = ri[threadIdx.x + s2];
                    if (ov > rv[threadIdx.x] ||
                        (ov == rv[threadIdx.x] && oi < ri[threadIdx.x])) {
                        rv[threadIdx.x] = ov; ri[threadIdx.x] = oi;
                    }
                }
                __syncthreads();
            }
            if (threadIdx.x == 0) {
                int idx = ri[0];
                atomicOr(&mask32[(size_t)b * LL + idx], 1u << i);
                sm[idx] = -INFINITY;
            }
            __syncthreads();
        }
    }
}

// ---------------------------------------------------------------- K3: WMMA sum + conflict resolve
// Block = 128 threads = 4 waves; each wave handles 16 anchors via one
// v_wmma_f32_16x16x32_f16: D = ones(16x32) x maskpos(32x16). With an all-ones
// A-matrix the sum is invariant to the K-permutation of the B layout.
__global__ void __launch_bounds__(128)
k3_final(const float* __restrict__ iou,
         const float* __restrict__ anchor_points,
         const float* __restrict__ gt_bboxes,
         const float* __restrict__ pad_mask,
         unsigned*    __restrict__ mask32,
         int*         __restrict__ agt) {
    __shared__ float s_gt[NGT * 4];           // 128 floats
    __shared__ float s_pad[NGT];
    int b  = blockIdx.y;
    int l0 = blockIdx.x * 64;
    int t  = threadIdx.x;

    // CDNA5 async copy: stage gt boxes + pad mask into LDS (ASYNCcnt path).
    {
        unsigned lds = (unsigned)(uintptr_t)(&s_gt[t]);
        unsigned long long ga =
            (unsigned long long)(uintptr_t)(gt_bboxes + (size_t)b * NGT * 4 + t);
        asm volatile("global_load_async_to_lds_b32 %0, %1, off"
                     :: "v"(lds), "v"(ga) : "memory");
        if (t < NGT) {                        // whole wave 0: no intra-wave divergence
            unsigned lds2 = (unsigned)(uintptr_t)(&s_pad[t]);
            unsigned long long ga2 =
                (unsigned long long)(uintptr_t)(pad_mask + (size_t)b * NGT + t);
            asm volatile("global_load_async_to_lds_b32 %0, %1, off"
                         :: "v"(lds2), "v"(ga2) : "memory");
        }
    }
    asm volatile("s_wait_asynccnt 0x0" ::: "memory");
    __syncthreads();

    int wave = t >> 5;
    int lane = t & 31;
    int j    = lane & 15;
    int l    = l0 + wave * 16 + j;
    int lc   = (l < LL) ? l : (LL - 1);       // clamp so EXEC stays all-ones

    unsigned topk = mask32[(size_t)b * LL + lc];
    float ax = anchor_points[lc * 2], ay = anchor_points[lc * 2 + 1];

    int ibase = lane & 16;                    // lanes 0-15: gts 0..15; 16-31: gts 16..31
    v16h bm = {};
    unsigned my16 = 0;
    for (int h = 0; h < 16; ++h) {
        int i = ibase + h;
        float x0 = s_gt[i*4], y0 = s_gt[i*4+1], x1 = s_gt[i*4+2], y1 = s_gt[i*4+3];
        float dmin = fminf(fminf(ax - x0, ay - y0), fminf(x1 - ax, y1 - ay));
        bool bit = ((topk >> i) & 1u) && (dmin > EPSF) && (s_pad[i] > 0.f);
        bm[h] = bit ? (_Float16)1.0f : (_Float16)0.0f;
        if (bit) my16 |= (1u << h);
    }
    v16h av;
    for (int h = 0; h < 16; ++h) av[h] = (_Float16)1.0f;
    v8f c = {};
    c = __builtin_amdgcn_wmma_f32_16x16x32_f16(
            false, av, false, bm, (short)0, c, false, false);
    int psum = (int)(c[0] + 0.5f);            // mask_positive_sum for anchor (lane&15)

    unsigned other = (unsigned)__shfl_xor((int)my16, 16, 32);

    if (lane < 16 && l < LL) {
        unsigned full = my16 | (other << 16); // full 32-gt positive mask
        float bvi = -INFINITY; int bidx = 0;  // argmax_i iou (first occurrence)
        for (int i = 0; i < NGT; ++i) {
            float v = iou[((size_t)b * NGT + i) * LL + l];
            if (v > bvi) { bvi = v; bidx = i; }
        }
        unsigned finalm = (psum > 1) ? (1u << bidx) : full;
        int gi = finalm ? (__ffs(finalm) - 1) : 0;   // argmax of 0/1 column
        mask32[(size_t)b * LL + l] = finalm;
        agt[(size_t)b * LL + l]    = gi;
    }
}

// ---------------------------------------------------------------- K4: per-instance maxima
__global__ void k4_permax(const float* __restrict__ iou,
                          const float* __restrict__ pred_scores,
                          const int*   __restrict__ gt_labels,
                          const unsigned* __restrict__ mask32,
                          float* __restrict__ maxm,
                          float* __restrict__ maxiou) {
    __shared__ float rm[256], rj[256];
    int bi = blockIdx.x;
    int b  = bi / NGT;
    int i  = bi % NGT;
    int lbl = gt_labels[bi];
    float mm = 0.f, mi = 0.f;
    for (int l = threadIdx.x; l < LL; l += blockDim.x) {
        unsigned msk = mask32[(size_t)b * LL + l];
        if ((msk >> i) & 1u) {
            float v  = iou[(size_t)bi * LL + l];
            float s  = pred_scores[((size_t)b * LL + l) * CC + lbl];
            float v2 = v * v;
            mm = fmaxf(mm, s * v2 * v2 * v2);
            mi = fmaxf(mi, v);
        }
    }
    rm[threadIdx.x] = mm; rj[threadIdx.x] = mi;
    __syncthreads();
    for (int s2 = 128; s2 > 0; s2 >>= 1) {
        if ((int)threadIdx.x < s2) {
            rm[threadIdx.x] = fmaxf(rm[threadIdx.x], rm[threadIdx.x + s2]);
            rj[threadIdx.x] = fmaxf(rj[threadIdx.x], rj[threadIdx.x + s2]);
        }
        __syncthreads();
    }
    if (threadIdx.x == 0) { maxm[bi] = rm[0]; maxiou[bi] = rj[0]; }
}

// ---------------------------------------------------------------- K5a: scalar outputs
__global__ void k5a_out(const float* __restrict__ iou,
                        const float* __restrict__ pred_scores,
                        const int*   __restrict__ gt_labels,
                        const float* __restrict__ gt_bboxes,
                        const float* __restrict__ gt_rot,
                        const unsigned* __restrict__ mask32,
                        const int*   __restrict__ agt,
                        const float* __restrict__ maxm,
                        const float* __restrict__ maxiou,
                        const int*   __restrict__ bgp,
                        float* __restrict__ out) {
    size_t idx = (size_t)blockIdx.x * 256 + threadIdx.x;
    const size_t BL = (size_t)BB * LL;
    if (idx >= BL) return;
    int b = (int)(idx / LL);
    int l = (int)(idx % LL);
    int bg = *bgp;
    unsigned msk = mask32[idx];
    int gi = agt[idx];
    int gidx = b * NGT + gi;
    int lblA = msk ? gt_labels[gidx] : bg;

    float am = 0.f;
    unsigned m2 = msk;
    while (m2) {
        int i = __ffs(m2) - 1; m2 &= m2 - 1;
        int bi = b * NGT + i;
        float v  = iou[(size_t)bi * LL + l];
        float s  = pred_scores[idx * CC + gt_labels[bi]];
        float v2 = v * v;
        float met = s * v2 * v2 * v2;
        am = fmaxf(am, met / (maxm[bi] + EPSF) * maxiou[bi]);
    }
    out[idx] = (float)lblA;                                   // labels
    float4 bb = *(const float4*)(gt_bboxes + (size_t)gidx * 4);
    *(float4*)(out + BL + idx * 4) = bb;                      // bboxes
    for (int cc = 0; cc < CC; ++cc) {                         // scores (keep != bg)
        int cls = (cc < bg) ? cc : cc + 1;
        out[BL * 5 + idx * CC + cc] = (lblA == cls) ? am : 0.f;
    }
    const float* rs = gt_rot + (size_t)gidx * 9;              // rotations
    float* rd = out + BL * 414 + idx * 9;
    for (int r = 0; r < 9; ++r) rd[r] = rs[r];
    out[BL * 423 + idx] = (float)(gi + b * NGT);              // assigned_gt_index
}

// ---------------------------------------------------------------- K5b: poses + vertices (streaming)
__global__ void k5b_gather(const float* __restrict__ gt_poses,
                           const float* __restrict__ gt_verts,
                           const int*   __restrict__ agt,
                           float* __restrict__ out) {
    const size_t BL = (size_t)BB * LL;
    const int    PER = 2 * KPT * 3;           // 408 floats / anchor
    size_t idx = (size_t)blockIdx.x * 256 + threadIdx.x;
    size_t total = BL * (size_t)PER;
    if (idx >= total) return;
    size_t a = idx / PER;
    int    r = (int)(idx % PER);
    int    b = (int)(a / LL);
    int    gi = agt[a];
    size_t g = ((size_t)b * NGT + gi) * (size_t)(KPT * 3);
    if (r < KPT * 3) out[BL * 6   + a * (KPT * 3) + r]            = gt_poses[g + r];
    else             out[BL * 210 + a * (KPT * 3) + (r - KPT*3)]  = gt_verts[g + r - KPT*3];
}

// ---------------------------------------------------------------- launcher
extern "C" void kernel_launch(void* const* d_in, const int* in_sizes, int n_in,
                              void* d_out, int out_size, void* d_ws, size_t ws_size,
                              hipStream_t stream) {
    const float* pred_scores   = (const float*)d_in[0];
    const float* pred_bboxes   = (const float*)d_in[1];
    const float* anchor_points = (const float*)d_in[2];
    const int*   gt_labels     = (const int*)  d_in[3];
    const float* gt_bboxes     = (const float*)d_in[4];
    const float* gt_poses      = (const float*)d_in[5];
    const float* gt_vertices   = (const float*)d_in[6];
    const float* gt_rotations  = (const float*)d_in[7];
    const float* pad_gt_mask   = (const float*)d_in[8];
    const int*   bg_index      = (const int*)  d_in[9];
    float* out = (float*)d_out;

    const size_t BL = (size_t)BB * LL;
    char* ws = (char*)d_ws;
    size_t off = 0;
    float*    iou    = (float*)(ws + off);  off += (size_t)BB * NGT * LL * 4;
    unsigned* mask32 = (unsigned*)(ws + off); off += BL * 4;
    int*      agt    = (int*)(ws + off);    off += BL * 4;
    float*    maxm   = (float*)(ws + off);  off += (size_t)BB * NGT * 4;
    float*    maxiou = (float*)(ws + off);  off += (size_t)BB * NGT * 4;

    k0_zero<<<(int)((BL + 255) / 256), 256, 0, stream>>>(mask32, (int)BL);

    dim3 g1((LL + 255) / 256, BB * NGT);
    k1_iou<<<g1, 256, 0, stream>>>(gt_bboxes, pred_bboxes, iou);

    k2_topk<<<BB * NGT, 256, 0, stream>>>(iou, pred_scores, anchor_points,
                                          gt_bboxes, pad_gt_mask, gt_labels, mask32);

    dim3 g3((LL + 63) / 64, BB);
    k3_final<<<g3, 128, 0, stream>>>(iou, anchor_points, gt_bboxes,
                                     pad_gt_mask, mask32, agt);

    k4_permax<<<BB * NGT, 256, 0, stream>>>(iou, pred_scores, gt_labels,
                                            mask32, maxm, maxiou);

    k5a_out<<<(int)((BL + 255) / 256), 256, 0, stream>>>(
        iou, pred_scores, gt_labels, gt_bboxes, gt_rotations,
        mask32, agt, maxm, maxiou, bg_index, out);

    size_t tot5b = BL * (size_t)(2 * KPT * 3);
    k5b_gather<<<(int)((tot5b + 255) / 256), 256, 0, stream>>>(
        gt_poses, gt_vertices, agt, out);
}
